// SparseLinear_6914897347207
// MI455X (gfx1250) — compile-verified, matrix operand order checked by reading, not run on hardware
//
#include <hip/hip_runtime.h>
#include <hip/hip_bf16.h>

typedef __attribute__((ext_vector_type(16))) __bf16 v16bf;
typedef __attribute__((ext_vector_type(8)))  __bf16 v8bf;
typedef __attribute__((ext_vector_type(8)))  float  v8f;

// ---------- fp32 -> bf16 (round to nearest even) ----------
__device__ __forceinline__ unsigned int f2bf(float f) {
    unsigned int u = __builtin_bit_cast(unsigned int, f);
    unsigned int r = u + 0x7FFFu + ((u >> 16) & 1u);
    return r >> 16;
}

// Convert n fp32 values to bf16 (n must be a multiple of 8 — true here).
__global__ __launch_bounds__(256) void cvt_bf16_kernel(
    const float* __restrict__ src, unsigned short* __restrict__ dst, long long n)
{
    long long i = ((long long)blockIdx.x * blockDim.x + threadIdx.x) * 8;
    if (i + 8 > n) return;
    const float4* s = reinterpret_cast<const float4*>(src + i);
    float4 f0 = s[0];
    float4 f1 = s[1];
    uint4 o;
    o.x = f2bf(f0.x) | (f2bf(f0.y) << 16);
    o.y = f2bf(f0.z) | (f2bf(f0.w) << 16);
    o.z = f2bf(f1.x) | (f2bf(f1.y) << 16);
    o.w = f2bf(f1.z) | (f2bf(f1.w) << 16);
    *reinterpret_cast<uint4*>(dst + i) = o;
}

// Build per-N-block list of active K-blocks by sampling the (N,K) mask at
// block corners (mask is constant within each 32x32 block).
__global__ void blocklist_kernel(const float* __restrict__ mask,
                                 int* __restrict__ counts, int* __restrict__ idx,
                                 int NB, int KB, long long K)
{
    int nb = blockIdx.x * blockDim.x + threadIdx.x;
    if (nb >= NB) return;
    long long rowoff = (long long)nb * 32 * K;
    int c = 0;
    for (int kb = 0; kb < KB; ++kb) {
        if (mask[rowoff + (long long)kb * 32] != 0.0f)
            idx[nb * KB + c++] = kb;
    }
    counts[nb] = c;
}

// Load one 16x32 bf16 WMMA fragment at byte offset `off` from uniform base.
// ISA 16-bit A layout: lanes 0-15 row M=l, K=0..7 then 16..23;
//                      lanes 16-31 row M=l-16, K=8..15 then 24..31.
// (the k-half select is folded into the precomputed lane byte offset)
__device__ __forceinline__ v16bf load_frag(const char* __restrict__ base, unsigned off)
{
    v8bf lo = *reinterpret_cast<const v8bf*>(base + off);        // K = k0 .. k0+7
    v8bf hi = *reinterpret_cast<const v8bf*>(base + off + 32);   // K = k0+16 .. k0+23
    v16bf f;
#pragma unroll
    for (int i = 0; i < 8; ++i) { f[i] = lo[i]; f[i + 8] = hi[i]; }
    return f;
}

__device__ __forceinline__ v8f wmma_bf16(v16bf a, v16bf b, v8f c)
{
    return __builtin_amdgcn_wmma_f32_16x16x32_bf16(false, a, false, b,
                                                   (short)0, c, false, false);
}

struct FragSet { v16bf a[4]; v16bf b[2]; };

__device__ __forceinline__ void load_set(const char* __restrict__ Ab,
                                         const char* __restrict__ Wb,
                                         const unsigned offA[4], const unsigned offB[2],
                                         unsigned kbytes, FragSet& s)
{
#pragma unroll
    for (int ai = 0; ai < 4; ++ai) s.a[ai] = load_frag(Ab, offA[ai] + kbytes);
#pragma unroll
    for (int bi = 0; bi < 2; ++bi) s.b[bi] = load_frag(Wb, offB[bi] + kbytes);
}

__device__ __forceinline__ void compute_set(const FragSet& s, v8f acc[4][2])
{
#pragma unroll
    for (int ai = 0; ai < 4; ++ai) {
        acc[ai][0] = wmma_bf16(s.a[ai], s.b[0], acc[ai][0]);
        acc[ai][1] = wmma_bf16(s.a[ai], s.b[1], acc[ai][1]);
    }
}

// Block-sparse GEMM: C[M,N] = Abf[M,K] * Wbf[N,K]^T + bias, skipping inactive
// 32-wide K blocks. Grid: (N/32, M/512). Block: 256 threads = 8 wave32s.
// Wave w computes a 64(M) x 32(N) tile: 4 A-fragments x 2 B-fragments,
// 8 WMMAs per active k-block, explicitly double-buffered so the next
// k-block's loads are in flight while the current WMMAs execute.
__global__ __launch_bounds__(256) void sparse_gemm_kernel(
    const __bf16* __restrict__ Abf, const __bf16* __restrict__ Wbf,
    const float* __restrict__ bias, const int* __restrict__ counts,
    const int* __restrict__ idx, float* __restrict__ out,
    int Ki, int Ni, int KB)
{
    const int lane = threadIdx.x & 31;
    const int wave = threadIdx.x >> 5;
    const int nb   = blockIdx.x;
    const int n0   = nb * 32;
    const int m0   = blockIdx.y * 512 + wave * 64;

    const unsigned Ku   = (unsigned)Ki;
    const unsigned ksel = (lane & 16) ? 8u : 0u;   // k-half select per ISA layout
    const unsigned lrow = (unsigned)(lane & 15);

    // Lane-constant BYTE offsets: buffers < 2^31 bytes -> 32-bit voffset from
    // a uniform saddr base, avoiding 64-bit VALU address arithmetic.
    unsigned offA[4], offB[2];
#pragma unroll
    for (int ai = 0; ai < 4; ++ai)
        offA[ai] = (((unsigned)m0 + 16u * ai + lrow) * Ku + ksel) * 2u;
#pragma unroll
    for (int bi = 0; bi < 2; ++bi)
        offB[bi] = (((unsigned)n0 + 16u * bi + lrow) * Ku + ksel) * 2u;

    const char* Ab = (const char*)Abf;
    const char* Wb = (const char*)Wbf;

    v8f acc[4][2];
#pragma unroll
    for (int ai = 0; ai < 4; ++ai)
#pragma unroll
        for (int bi = 0; bi < 2; ++bi)
            acc[ai][bi] = (v8f){};

    const int  cnt = counts[nb];
    const int* lst = idx + nb * KB;

    if (cnt > 0) {
        FragSet s0, s1;
        load_set(Ab, Wb, offA, offB, (unsigned)lst[0] * 64u, s0);
        int i = 0;
        for (; i + 2 <= cnt; i += 2) {
            // prefetch set i+1 while computing set i
            load_set(Ab, Wb, offA, offB, (unsigned)lst[i + 1] * 64u, s1);
            compute_set(s0, acc);
            // prefetch set i+2 (clamped: unconditional straight-line code)
            const int nx = (i + 2 < cnt) ? (i + 2) : (cnt - 1);
            load_set(Ab, Wb, offA, offB, (unsigned)lst[nx] * 64u, s0);
            compute_set(s1, acc);
        }
        if (i < cnt)           // odd tail: its fragments are already in s0
            compute_set(s0, acc);
    }

    // Epilogue. C/D layout: VGPR g, lanes 0-15 -> M = g, lanes 16-31 -> M = g+8;
    // N = lane % 16 (per 16-wide half).
    const unsigned Nu    = (unsigned)Ni;
    const unsigned col0  = (unsigned)n0 + lrow;
    const float    bias0 = bias[col0];
    const float    bias1 = bias[col0 + 16];
    const unsigned rsub  = ((lane >> 4) & 1) * 8u;
#pragma unroll
    for (int ai = 0; ai < 4; ++ai) {
        const unsigned rbase = (unsigned)m0 + 16u * ai + rsub;
#pragma unroll
        for (int g = 0; g < 8; ++g) {
            const unsigned r = rbase + g;
            out[r * Nu + col0]      = acc[ai][0][g] + bias0;
            out[r * Nu + col0 + 16] = acc[ai][1][g] + bias1;
        }
    }
}

extern "C" void kernel_launch(void* const* d_in, const int* in_sizes, int n_in,
                              void* d_out, int out_size, void* d_ws, size_t ws_size,
                              hipStream_t stream)
{
    const float* A    = (const float*)d_in[0];   // (M, K)
    const float* W    = (const float*)d_in[1];   // (N, K)
    const float* bias = (const float*)d_in[2];   // (N,)
    const float* mask = (const float*)d_in[3];   // (N, K)
    float* out = (float*)d_out;                  // (M, N)

    const long long N = in_sizes[2];
    const long long K = (long long)in_sizes[1] / N;
    const long long M = (long long)in_sizes[0] / K;
    const int NB = (int)(N / 32);
    const int KB = (int)(K / 32);

    // Workspace layout: [A bf16][W bf16][counts][idx]
    size_t offA   = 0;
    size_t offW   = offA + (size_t)(M * K) * sizeof(unsigned short);
    size_t offCnt = (offW + (size_t)(N * K) * sizeof(unsigned short) + 255) & ~(size_t)255;
    size_t offIdx = (offCnt + (size_t)NB * sizeof(int) + 255) & ~(size_t)255;

    unsigned short* Abf = (unsigned short*)((char*)d_ws + offA);
    unsigned short* Wbf = (unsigned short*)((char*)d_ws + offW);
    int* counts = (int*)((char*)d_ws + offCnt);
    int* idx    = (int*)((char*)d_ws + offIdx);

    // 1) A -> bf16
    {
        long long n = M * K;
        int blocks = (int)((n / 8 + 255) / 256);
        cvt_bf16_kernel<<<blocks, 256, 0, stream>>>(A, Abf, n);
    }
    // 2) W -> bf16 (no mask multiply needed: inactive blocks are skipped)
    {
        long long n = N * K;
        int blocks = (int)((n / 8 + 255) / 256);
        cvt_bf16_kernel<<<blocks, 256, 0, stream>>>(W, Wbf, n);
    }
    // 3) Active-block lists
    {
        int blocks = (NB + 127) / 128;
        blocklist_kernel<<<blocks, 128, 0, stream>>>(mask, counts, idx, NB, KB, K);
    }
    // 4) Block-sparse WMMA GEMM
    {
        dim3 grid((unsigned)NB, (unsigned)(M / 512));
        sparse_gemm_kernel<<<grid, 256, 0, stream>>>(
            (const __bf16*)Abf, (const __bf16*)Wbf, bias, counts, idx, out,
            (int)K, (int)N, KB);
    }
}
